// Model_2250562863841
// MI455X (gfx1250) — matrix operand order repaired and checked by reading, not compiled, not run on hardware
//
#include <hip/hip_runtime.h>

typedef __attribute__((ext_vector_type(16))) __bf16 v16bf;
typedef __attribute__((ext_vector_type(8)))  __bf16 v8bf;
typedef __attribute__((ext_vector_type(8)))  float  v8f;

#define BS_ 32
#define SZ_ 50
#define NL_ 100
#define E_  256
#define DENSE_ 384
#define NH_ 16
#define NC_ 250

__device__ __forceinline__ float mishf(float x) {
  float sp = (x > 20.f) ? x : log1pf(expf(x));
  return x * tanhf(sp);
}

// ---------------- generic f32 -> bf16 ----------------
__global__ void k_f32_to_bf16(const float* __restrict__ src, __bf16* __restrict__ dst, int n) {
  int i = blockIdx.x * 256 + threadIdx.x;
  if (i < n) dst[i] = (__bf16)src[i];
}

// ---------------- layernorm over last dim (block per row) ----------------
__global__ void k_layernorm(const float* __restrict__ X, const float* __restrict__ res,
                            const float* __restrict__ g, const float* __restrict__ bta,
                            float eps, int N, float* __restrict__ outF, __bf16* __restrict__ outB) {
  int row = blockIdx.x;
  int tid = threadIdx.x;
  __shared__ float r1[256], r2[256];
  const float* xr = X + (size_t)row * N;
  const float* rr = res ? res + (size_t)row * N : nullptr;
  float s = 0.f, s2 = 0.f;
  for (int c = tid; c < N; c += 256) {
    float x = xr[c]; if (rr) x += rr[c];
    s += x; s2 += x * x;
  }
  r1[tid] = s; r2[tid] = s2; __syncthreads();
  for (int o = 128; o > 0; o >>= 1) {
    if (tid < o) { r1[tid] += r1[tid + o]; r2[tid] += r2[tid + o]; }
    __syncthreads();
  }
  float mean = r1[0] / (float)N;
  float var  = r2[0] / (float)N - mean * mean;
  float rstd = rsqrtf(var + eps);
  for (int c = tid; c < N; c += 256) {
    float x = xr[c]; if (rr) x += rr[c];
    float y = (x - mean) * rstd * g[c] + bta[c];
    outF[(size_t)row * N + c] = y;
    if (outB) outB[(size_t)row * N + c] = (__bf16)y;
  }
}

// ---------------- small scalar proj: C[M,N] = A[M,K] @ W[:, woff:woff+K]^T ----------------
__global__ void k_small_proj(const float* __restrict__ A, int lda,
                             const float* __restrict__ W, int ldw, int woff,
                             float* __restrict__ C, int M, int N, int K) {
  int idx = blockIdx.x * 256 + threadIdx.x;
  if (idx >= M * N) return;
  int r = idx / N, c = idx % N;
  const float* a = A + (size_t)r * lda;
  const float* w = W + (size_t)c * ldw + woff;
  float acc = 0.f;
  for (int t = 0; t < K; ++t) acc += a[t] * w[t];
  C[idx] = acc;
}

// ---------------- fold pos_dense into dense W3: Mcomb[19,256], c0[256] ----------------
__global__ void k_pos_combine(const float* __restrict__ posW, const float* __restrict__ posB,
                              const float* __restrict__ dW, const float* __restrict__ dB,
                              float* __restrict__ Mcomb, float* __restrict__ c0) {
  int idx = blockIdx.x * 256 + threadIdx.x;
  if (idx < 19 * 256) {
    int i = idx / 256, c = idx % 256;
    float acc = 0.f;
    for (int t = 0; t < 256; ++t) acc += posW[t * 19 + i] * dW[c * 768 + 512 + t];
    Mcomb[i * 256 + c] = acc;
  } else if (idx < 19 * 256 + 256) {
    int c = idx - 19 * 256;
    float acc = dB[c];
    for (int t = 0; t < 256; ++t) acc += posB[t] * dW[c * 768 + 512 + t];
    c0[c] = acc;
  }
}

// ---------------- temporal conv 3->8->16, k=5, same pad; emits pos_in[...,19] ----------------
__global__ void k_conv(const float* __restrict__ x, const float* __restrict__ w1,
                       const float* __restrict__ w2, float* __restrict__ posIn) {
  int bl = blockIdx.x;         // b*NL + l
  int b = bl / NL_, l = bl % NL_;
  int tid = threadIdx.x;       // 64 threads
  __shared__ float in0[3][52];
  __shared__ float c1[8][52];
  __shared__ float c2[16][52];
  for (int idx = tid; idx < 3 * 52; idx += 64) {
    int ch = idx / 52, t = idx % 52;
    in0[ch][t] = (t < SZ_) ? x[(((size_t)(b * SZ_ + t)) * 5 + 1 + ch) * NL_ + l] : 0.f;
  }
  __syncthreads();
  for (int idx = tid; idx < 8 * 52; idx += 64) {
    int co = idx / 52, t = idx % 52;
    float acc = 0.f;
    for (int ci = 0; ci < 3; ++ci)
      for (int kk = 0; kk < 5; ++kk) {
        int tt = t + kk - 2;
        if (tt >= 0 && tt < 52) acc += in0[ci][tt] * w1[(co * 3 + ci) * 5 + kk];
      }
    c1[co][t] = acc;
  }
  __syncthreads();
  for (int idx = tid; idx < 16 * 52; idx += 64) {
    int co = idx / 52, t = idx % 52;
    float acc = 0.f;
    for (int ci = 0; ci < 8; ++ci)
      for (int kk = 0; kk < 5; ++kk) {
        int tt = t + kk - 2;
        if (tt >= 0 && tt < 52) acc += c1[ci][tt] * w2[(co * 8 + ci) * 5 + kk];
      }
    c2[co][t] = acc;
  }
  __syncthreads();
  for (int idx = tid; idx < SZ_ * 19; idx += 64) {
    int s = idx / 19, i = idx % 19;
    float v = (i < 3) ? in0[i][s] : c2[i - 3][s];
    posIn[((size_t)((b * SZ_ + s) * NL_ + l)) * 19 + i] = v;
  }
}

// ---------------- fused fts: gathers + 19-dot, bf16 out ----------------
__global__ void k_fts(const float* __restrict__ x, const float* __restrict__ posIn,
                      const float* __restrict__ projTE, const float* __restrict__ projLE,
                      const float* __restrict__ Mcomb, const float* __restrict__ c0,
                      __bf16* __restrict__ fts) {
  int l = blockIdx.x, s = blockIdx.y, b = blockIdx.z;
  int c = threadIdx.x;  // 256
  __shared__ float p[19];
  __shared__ int ti, li;
  size_t row = (size_t)(b * SZ_ + s) * NL_ + l;
  if (c < 19) p[c] = posIn[row * 19 + c];
  if (c == 32) {
    int t = (int)x[(((size_t)(b * SZ_ + s)) * 5 + 0) * NL_ + l];
    ti = min(max(t, 0), 8);
  }
  if (c == 33) {
    int t = (int)x[(((size_t)(b * SZ_ + s)) * 5 + 4) * NL_ + l];
    li = min(max(t, 0), 100);
  }
  __syncthreads();
  float acc = projTE[ti * E_ + c] + projLE[li * E_ + c] + c0[c];
#pragma unroll
  for (int i = 0; i < 19; ++i) acc += p[i] * Mcomb[i * E_ + c];
  fts[row * E_ + c] = (__bf16)acc;
}

// ---------------- WMMA bf16 GEMM: C[M,N] = A[M,K](lda) @ B[N,K](ldb)^T + bias ----------------
// block = 8 waves; wave w owns rows [blockIdx.y*256 + w*32, +32), cols [blockIdx.x*64, +64)
// Branchless clamped loads; single fragment bank with unroll-by-2 so the scheduler can
// interleave loads/WMMAs itself while peak live VGPRs stay ~130 (no spills).
__device__ __forceinline__ v16bf ldfrag(const __bf16* p0, const __bf16* p1) {
  v8bf lo = *(const v8bf*)p0;
  v8bf hi = *(const v8bf*)p1;
  v16bf r;
#pragma unroll
  for (int i = 0; i < 8; ++i) { r[i] = lo[i]; r[8 + i] = hi[i]; }
  return r;
}

#define WMMA1(ACC, AA, BB)                                                        \
  ACC = __builtin_amdgcn_wmma_f32_16x16x32_bf16(false, AA, false, BB, (short)0,   \
                                                ACC, false, false)

__global__ void k_gemm_bf16(const __bf16* __restrict__ A, int lda,
                            const __bf16* __restrict__ B, int ldb,
                            const float* __restrict__ bias,
                            float* __restrict__ C, int ldc,
                            int M, int N, int K) {
  const int lane  = threadIdx.x & 31;
  const int wave  = threadIdx.x >> 5;
  const int tileM = blockIdx.y * 256 + wave * 32;
  const int tileN = blockIdx.x * 64;
  const int nlane = lane & 15;
  const int half  = lane >> 4;

  // clamped base pointers (branchless; out-of-range lanes masked at store time)
  const __bf16* aP0 = A + (size_t)min(tileM + nlane, M - 1) * lda + 16 * half;
  const __bf16* aP1 = A + (size_t)min(tileM + 16 + nlane, M - 1) * lda + 16 * half;
  const __bf16* bP0 = B + (size_t)min(tileN + nlane, N - 1) * ldb + 16 * half;
  const __bf16* bP1 = B + (size_t)min(tileN + 16 + nlane, N - 1) * ldb + 16 * half;
  const __bf16* bP2 = B + (size_t)min(tileN + 32 + nlane, N - 1) * ldb + 16 * half;
  const __bf16* bP3 = B + (size_t)min(tileN + 48 + nlane, N - 1) * ldb + 16 * half;

  v8f acc00 = {}, acc01 = {}, acc02 = {}, acc03 = {};
  v8f acc10 = {}, acc11 = {}, acc12 = {}, acc13 = {};

#pragma clang loop unroll_count(2)
  for (int k0 = 0; k0 < K; k0 += 32) {
    v16bf ax = ldfrag(aP0 + k0, aP0 + k0 + 8);
    v16bf ay = ldfrag(aP1 + k0, aP1 + k0 + 8);
    v16bf ba = ldfrag(bP0 + k0, bP0 + k0 + 8);
    v16bf bb = ldfrag(bP1 + k0, bP1 + k0 + 8);
    v16bf bc = ldfrag(bP2 + k0, bP2 + k0 + 8);
    v16bf bd = ldfrag(bP3 + k0, bP3 + k0 + 8);
    __builtin_prefetch(aP0 + k0 + 128, 0, 3);
    __builtin_prefetch(aP1 + k0 + 128, 0, 3);
    WMMA1(acc00, ax, ba); WMMA1(acc01, ax, bb);
    WMMA1(acc02, ax, bc); WMMA1(acc03, ax, bd);
    WMMA1(acc10, ay, ba); WMMA1(acc11, ay, bb);
    WMMA1(acc12, ay, bc); WMMA1(acc13, ay, bd);
  }

  v8f accArr[2][4] = {{acc00, acc01, acc02, acc03}, {acc10, acc11, acc12, acc13}};
#pragma unroll
  for (int m2 = 0; m2 < 2; ++m2)
#pragma unroll
    for (int j = 0; j < 4; ++j) {
      int col = tileN + j * 16 + nlane;
      if (col >= N) continue;
      float bv = bias ? bias[col] : 0.f;
#pragma unroll
      for (int r = 0; r < 8; ++r) {
        int row = tileM + m2 * 16 + 8 * half + r;
        if (row < M) C[(size_t)row * ldc + col] = accArr[m2][j][r] + bv;
      }
    }
}

// ---------------- BN (inference) + mish epilogue ----------------
__global__ void k_bn_mish(const float* __restrict__ X,
                          const float* __restrict__ g, const float* __restrict__ bta,
                          const float* __restrict__ mm, const float* __restrict__ vv,
                          int M, int N,
                          float* __restrict__ outF,
                          __bf16* __restrict__ outB, int ldOutB, int colOff) {
  int idx = blockIdx.x * 256 + threadIdx.x;
  if (idx >= M * N) return;
  int r = idx / N, c = idx % N;
  float y = (X[idx] - mm[c]) * rsqrtf(vv[c] + 1e-5f) * g[c] + bta[c];
  y = mishf(y);
  if (outF) outF[idx] = y;
  if (outB) outB[(size_t)r * ldOutB + colOff + c] = (__bf16)y;
}

__global__ void k_hand_max(const float* __restrict__ L, const float* __restrict__ R,
                           __bf16* __restrict__ outB, int ldOut, int colOff, int M, int N) {
  int idx = blockIdx.x * 256 + threadIdx.x;
  if (idx >= M * N) return;
  int r = idx / N, c = idx % N;
  outB[(size_t)r * ldOut + colOff + c] = (__bf16)fmaxf(L[idx], R[idx]);
}

__global__ void k_mish_add_bf16(const float* __restrict__ X, const float* __restrict__ Y,
                                __bf16* __restrict__ outB, int n) {
  int i = blockIdx.x * 256 + threadIdx.x;
  if (i < n) outB[i] = (__bf16)(mishf(X[i]) + Y[i]);
}

// ---------------- disentangled attention (c2p + p2c), one block per (b, head) ----------------
__global__ void k_attention(const float* __restrict__ q, const float* __restrict__ k,
                            const float* __restrict__ v,
                            const float* __restrict__ posk, const float* __restrict__ posq,
                            __bf16* __restrict__ ctx, int hd, int d) {
  int b = blockIdx.x / NH_;
  int h = blockIdx.x % NH_;
  int tid = threadIdx.x;
  __shared__ float qs[SZ_ * 48], ks[SZ_ * 48], vs[SZ_ * 48];
  __shared__ float sc[SZ_ * SZ_];
  int rowBase = b * SZ_;
  for (int idx = tid; idx < SZ_ * d; idx += 256) {
    int i = idx / d, dd = idx % d;
    size_t off = (size_t)(rowBase + i) * hd + h * d + dd;
    qs[i * d + dd] = q[off];
    ks[i * d + dd] = k[off];
    vs[i * d + dd] = v[off];
  }
  __syncthreads();
  float scale = rsqrtf(3.0f * (float)d);
  for (int idx = tid; idx < SZ_ * SZ_; idx += 256) {
    int i = idx / SZ_, j = idx % SZ_;
    int r = i - j + 50; r = min(max(r, 0), 99);
    const float* pk = posk + (size_t)r * hd + h * d;
    const float* pq = posq + (size_t)r * hd + h * d;
    float s = 0.f;
    for (int dd = 0; dd < d; ++dd) {
      float qv = qs[i * d + dd], kv = ks[j * d + dd];
      s += qv * kv + qv * pk[dd] + kv * pq[dd];
    }
    sc[idx] = s * scale;
  }
  __syncthreads();
  if (tid < SZ_) {
    int i = tid;
    float mx = -1e30f;
    for (int j = 0; j < SZ_; ++j) mx = fmaxf(mx, sc[i * SZ_ + j]);
    float sum = 0.f;
    for (int j = 0; j < SZ_; ++j) { float e = expf(sc[i * SZ_ + j] - mx); sc[i * SZ_ + j] = e; sum += e; }
    float inv = 1.f / sum;
    for (int j = 0; j < SZ_; ++j) sc[i * SZ_ + j] *= inv;
  }
  __syncthreads();
  for (int idx = tid; idx < SZ_ * d; idx += 256) {
    int i = idx / d, dd = idx % d;
    float a = 0.f;
    for (int j = 0; j < SZ_; ++j) a += sc[i * SZ_ + j] * vs[j * d + dd];
    ctx[(size_t)(rowBase + i) * hd + h * d + dd] = (__bf16)a;
  }
}

__global__ void k_mean_pool(const float* __restrict__ H, __bf16* __restrict__ outB, int D) {
  int idx = blockIdx.x * 256 + threadIdx.x;
  if (idx >= BS_ * D) return;
  int b = idx / D, c = idx % D;
  float acc = 0.f;
  for (int s = 0; s < SZ_; ++s) acc += H[(size_t)(b * SZ_ + s) * D + c];
  outB[idx] = (__bf16)(acc / (float)SZ_);
}

// =====================================================================================
extern "C" void kernel_launch(void* const* d_in, const int* in_sizes, int n_in,
                              void* d_out, int out_size, void* d_ws, size_t ws_size,
                              hipStream_t stream) {
  (void)in_sizes; (void)n_in; (void)out_size; (void)ws_size;
  auto F = [&](int i) { return (const float*)d_in[i]; };

  char* wsB = (char*)d_ws;
  size_t off = 0;
  auto allocF = [&](size_t n) { float* p = (float*)(wsB + off); off += ((n * 4 + 255) / 256) * 256; return p; };
  auto allocH = [&](size_t n) { __bf16* p = (__bf16*)(wsB + off); off += ((n * 2 + 255) / 256) * 256; return p; };
  auto cvt = [&](int idx, size_t n) {
    __bf16* p = allocH(n);
    k_f32_to_bf16<<<(int)((n + 255) / 256), 256, 0, stream>>>(F(idx), p, (int)n);
    return p;
  };
  auto gemm = [&](const __bf16* A, int lda, const __bf16* B, int ldb, const float* bias,
                  float* C, int ldc, int M, int N, int K) {
    dim3 grid((N + 63) / 64, (M + 255) / 256);
    k_gemm_bf16<<<grid, 256, 0, stream>>>(A, lda, B, ldb, bias, C, ldc, M, N, K);
  };

  const int R = BS_ * SZ_;  // 1600 token rows

  // ---- scratch ----
  float* lnTE   = allocF(9 * E_);
  float* lnLE   = allocF(101 * E_);
  float* projTE = allocF(9 * E_);
  float* projLE = allocF(101 * E_);
  float* Mcomb  = allocF(19 * E_);
  float* c0     = allocF(E_);
  float* posIn  = allocF((size_t)R * NL_ * 19);
  __bf16* ftsB  = allocH((size_t)R * NL_ * E_);
  float* gOut   = allocF((size_t)R * 768);
  float* Lbuf   = allocF((size_t)R * DENSE_);
  float* Rbuf   = allocF((size_t)R * DENSE_);
  __bf16* hcatB = allocH((size_t)R * 4 * DENSE_);
  float* hA     = allocF((size_t)R * 768);
  __bf16* hB    = allocH((size_t)R * 768);
  float* qF     = allocF((size_t)R * 768);
  float* kF     = allocF((size_t)R * 768);
  float* vF     = allocF((size_t)R * 768);
  float* bufT   = allocF((size_t)R * 768);
  float* attnF  = allocF((size_t)R * 768);
  __bf16* attnB = allocH((size_t)R * 768);
  __bf16* ctxB  = allocH((size_t)R * 768);
  __bf16* sumB  = allocH((size_t)R * 768);
  float* relF   = allocF(112 * 768);
  __bf16* relB  = allocH(112 * 768);
  float* poskF  = allocF(112 * 768);
  float* posqF  = allocF(112 * 768);
  __bf16* poolB = allocH(BS_ * 512);

  // ---- fold embeddings + pos_dense into precomputed tables ----
  k_layernorm<<<9, 256, 0, stream>>>(F(1), nullptr, F(3), F(4), 1e-5f, E_, lnTE, nullptr);
  k_layernorm<<<101, 256, 0, stream>>>(F(2), nullptr, F(5), F(6), 1e-5f, E_, lnLE, nullptr);
  k_small_proj<<<(9 * 256 + 255) / 256, 256, 0, stream>>>(lnTE, E_, F(11), 768, 0, projTE, 9, 256, 256);
  k_small_proj<<<(101 * 256 + 255) / 256, 256, 0, stream>>>(lnLE, E_, F(11), 768, 256, projLE, 101, 256, 256);
  k_pos_combine<<<20, 256, 0, stream>>>(F(9), F(10), F(11), F(12), Mcomb, c0);

  // ---- conv + fused fts ----
  k_conv<<<BS_ * NL_, 64, 0, stream>>>(F(0), F(7), F(8), posIn);
  {
    dim3 g(NL_, SZ_, BS_);
    k_fts<<<g, 256, 0, stream>>>(F(0), posIn, projTE, projLE, Mcomb, c0, ftsB);
  }

  // ---- branch MLPs (WMMA GEMM + BN + mish) ----
  const int FTSLD = NL_ * E_;  // 25600
  __bf16* wLeft = cvt(13, (size_t)DENSE_ * 21 * E_);
  __bf16* wRight = cvt(19, (size_t)DENSE_ * 21 * E_);
  __bf16* wLips = cvt(25, (size_t)DENSE_ * 21 * E_);
  __bf16* wFace = cvt(31, (size_t)DENSE_ * 25 * E_);
  __bf16* wFull = cvt(37, (size_t)DENSE_ * NL_ * E_);
  __bf16* wLmk  = cvt(43, (size_t)512 * 4 * DENSE_);

  int nEl = R * DENSE_;
  gemm(ftsB + 0, FTSLD, wLeft, 21 * E_, F(14), gOut, DENSE_, R, DENSE_, 21 * E_);
  k_bn_mish<<<(nEl + 255) / 256, 256, 0, stream>>>(gOut, F(15), F(16), F(17), F(18), R, DENSE_, Lbuf, nullptr, 0, 0);
  gemm(ftsB + 21 * E_, FTSLD, wRight, 21 * E_, F(20), gOut, DENSE_, R, DENSE_, 21 * E_);
  k_bn_mish<<<(nEl + 255) / 256, 256, 0, stream>>>(gOut, F(21), F(22), F(23), F(24), R, DENSE_, Rbuf, nullptr, 0, 0);
  k_hand_max<<<(nEl + 255) / 256, 256, 0, stream>>>(Lbuf, Rbuf, hcatB, 4 * DENSE_, DENSE_, R, DENSE_);
  gemm(ftsB + 42 * E_, FTSLD, wLips, 21 * E_, F(26), gOut, DENSE_, R, DENSE_, 21 * E_);
  k_bn_mish<<<(nEl + 255) / 256, 256, 0, stream>>>(gOut, F(27), F(28), F(29), F(30), R, DENSE_, nullptr, hcatB, 4 * DENSE_, 2 * DENSE_);
  gemm(ftsB + 63 * E_, FTSLD, wFace, 25 * E_, F(32), gOut, DENSE_, R, DENSE_, 25 * E_);
  k_bn_mish<<<(nEl + 255) / 256, 256, 0, stream>>>(gOut, F(33), F(34), F(35), F(36), R, DENSE_, nullptr, hcatB, 4 * DENSE_, 3 * DENSE_);
  gemm(ftsB, FTSLD, wFull, NL_ * E_, F(38), gOut, DENSE_, R, DENSE_, NL_ * E_);
  k_bn_mish<<<(nEl + 255) / 256, 256, 0, stream>>>(gOut, F(39), F(40), F(41), F(42), R, DENSE_, nullptr, hcatB, 4 * DENSE_, 0);
  gemm(hcatB, 4 * DENSE_, wLmk, 4 * DENSE_, F(44), gOut, 512, R, 512, 4 * DENSE_);
  k_bn_mish<<<(R * 512 + 255) / 256, 256, 0, stream>>>(gOut, F(45), F(46), F(47), F(48), R, 512, hA, hB, 512, 0);

  // ---- DeBERTa layers ----
  auto layer = [&](int base, int hd, int od) {
    int d = hd / NH_;
    k_layernorm<<<2 * 50, 256, 0, stream>>>(F(base + 0), nullptr, F(base + 1), F(base + 2),
                                            1e-7f, hd, relF, relB);
    __bf16* Wq = cvt(base + 3, (size_t)hd * hd);
    __bf16* Wk = cvt(base + 5, (size_t)hd * hd);
    __bf16* Wv = cvt(base + 7, (size_t)hd * hd);
    __bf16* Wo = cvt(base + 9, (size_t)hd * hd);
    __bf16* Wi = cvt(base + 13, (size_t)hd * hd);
    __bf16* Wout = cvt(base + 15, (size_t)od * hd);
    gemm(hB, hd, Wq, hd, F(base + 4), qF, hd, R, hd, hd);
    gemm(hB, hd, Wk, hd, F(base + 6), kF, hd, R, hd, hd);
    gemm(hB, hd, Wv, hd, F(base + 8), vF, hd, R, hd, hd);
    gemm(relB, hd, Wk, hd, F(base + 6), poskF, hd, 100, hd, hd);
    gemm(relB, hd, Wq, hd, F(base + 4), posqF, hd, 100, hd, hd);
    k_attention<<<BS_ * NH_, 256, 0, stream>>>(qF, kF, vF, poskF, posqF, ctxB, hd, d);
    gemm(ctxB, hd, Wo, hd, F(base + 10), bufT, hd, R, hd, hd);
    k_layernorm<<<R, 256, 0, stream>>>(bufT, hA, F(base + 11), F(base + 12), 1e-7f, hd, attnF, attnB);
    gemm(attnB, hd, Wi, hd, F(base + 14), bufT, hd, R, hd, hd);
    k_mish_add_bf16<<<(R * hd + 255) / 256, 256, 0, stream>>>(bufT, attnF, sumB, R * hd);
    gemm(sumB, hd, Wout, hd, F(base + 16), bufT, od, R, od, hd);
    k_layernorm<<<R, 256, 0, stream>>>(bufT, nullptr, F(base + 17), F(base + 18), 1e-7f, od, hA, hB);
  };
  layer(49, 512, 768);
  layer(68, 768, 768);
  layer(87, 768, 512);

  // ---- pool + logits ----
  k_mean_pool<<<(BS_ * 512 + 255) / 256, 256, 0, stream>>>(hA, poolB, 512);
  __bf16* wLg = cvt(106, (size_t)NC_ * 512);
  gemm(poolB, 512, wLg, 512, F(107), (float*)d_out, NC_, BS_, NC_, 512);
}
